// ClusterMemory_23519240913059
// MI455X (gfx1250) — compile-verified, matrix operand order checked by reading, not run on hardware
//
#include <hip/hip_runtime.h>
#include <hip/hip_bf16.h>

#define B_  256
#define D_  2048
#define N_  65536
#define NTILE 64                 // N columns per block
#define NBLK_ (N_ / NTILE)       // 1024 blocks
#define TEMP_INV 20.0f           // 1/0.05
#define MOM 0.2f
#define EPS_ 1e-12f

typedef __attribute__((ext_vector_type(16))) __bf16 v16bf;
typedef __attribute__((ext_vector_type(8)))  float  v8f;

union FragAB { v16bf v; uint4 q[2]; };

// ---------- helpers ----------
__device__ __forceinline__ float block_reduce_sum(float v, float* sbuf) {
    int t = threadIdx.x;
    sbuf[t] = v; __syncthreads();
    for (int s = 128; s > 0; s >>= 1) {
        if (t < s) sbuf[t] += sbuf[t + s];
        __syncthreads();
    }
    return sbuf[0];
}

__device__ __forceinline__ unsigned fkey(float f) {
    // monotonic increasing mapping float -> unsigned
    unsigned u = __float_as_uint(f);
    return (u & 0x80000000u) ? ~u : (u | 0x80000000u);
}

// ---------- 1) normalize inputs; emit f32 and bf16 copies ----------
__global__ void __launch_bounds__(256) normalize_inputs(
    const float* __restrict__ in, float* __restrict__ x, __bf16* __restrict__ xb)
{
    __shared__ float sred[256];
    const int b = blockIdx.x, t = threadIdx.x;
    float ss = 0.f;
    for (int i = t; i < D_; i += 256) {
        float v = in[(size_t)b * D_ + i];
        ss += v * v;
    }
    ss = block_reduce_sum(ss, sred);
    const float scale = 1.0f / fmaxf(sqrtf(ss), EPS_);
    for (int i = t; i < D_; i += 256) {
        float v = in[(size_t)b * D_ + i] * scale;
        x[(size_t)b * D_ + i]  = v;
        xb[(size_t)b * D_ + i] = (__bf16)v;
    }
}

// ---------- 2) WMMA GEMM + fused feature copy + fused online-softmax partials ----------
// Block = 64-wide N tile (64 feature rows) x all 256 batch rows. 8 waves,
// each wave owns M-tiles {w, w+8} x 4 N-tiles -> 8 f32 accumulators.
// K staged 64 at a time through LDS (f32 -> bf16), row stride padded to 72
// to spread the 16-row B-fragment reads across LDS banks.
__global__ void __launch_bounds__(256) gemm_fused(
    const float* __restrict__ feat, const __bf16* __restrict__ xb,
    float* __restrict__ pmax, float* __restrict__ psum,
    float* __restrict__ outfeat)
{
    __shared__ __align__(16) __bf16 sF[NTILE][72];   // 64 rows x 64 K (+8 pad)
    const int n0   = blockIdx.x * NTILE;
    const int tid  = threadIdx.x;
    const int lane = tid & 31;
    const int wv   = tid >> 5;          // 0..7
    const int half = lane >> 4;         // 0/1 -> K sub-block select
    const int lrow = lane & 15;         // M (for A) / N (for B) within tile

    const int srow = tid >> 2;          // staging: row 0..63
    const int scol = (tid & 3) * 16;    // staging: 16 floats per thread

    v8f acc0[4] = {{}, {}, {}, {}};
    v8f acc1[4] = {{}, {}, {}, {}};
    const uint4* xb4 = (const uint4*)xb;

    for (int kb = 0; kb < D_; kb += 64) {
        // ---- stage feature chunk: global f32 -> LDS bf16, plus passthrough copy ----
        const size_t gb = (size_t)(n0 + srow) * D_ + kb + scol;
#pragma unroll
        for (int u = 0; u < 4; ++u) {
            float4 fv = *(const float4*)&feat[gb + u * 4];
            outfeat[gb + u * 4 + 0] = fv.x;
            outfeat[gb + u * 4 + 1] = fv.y;
            outfeat[gb + u * 4 + 2] = fv.z;
            outfeat[gb + u * 4 + 3] = fv.w;
            sF[srow][scol + u * 4 + 0] = (__bf16)fv.x;
            sF[srow][scol + u * 4 + 1] = (__bf16)fv.y;
            sF[srow][scol + u * 4 + 2] = (__bf16)fv.z;
            sF[srow][scol + u * 4 + 3] = (__bf16)fv.w;
        }
        if (kb + 64 < D_)
            __builtin_prefetch(&feat[gb + 64], 0, 1);   // global_prefetch_b8
        __syncthreads();

        // ---- two K=32 WMMA steps per chunk, 4 N-tiles each ----
#pragma unroll
        for (int ks = 0; ks < 64; ks += 32) {
            FragAB a0, a1;
            // A fragments (bf16 16x32): half 0 -> K {0..7,16..23}, half 1 -> {8..15,24..31}
            const size_t r0 = (size_t)(wv * 16 + lrow) * D_ + kb + ks;
            a0.q[0] = xb4[(r0 + (size_t)half * 8) >> 3];
            a0.q[1] = xb4[(r0 + 16 + (size_t)half * 8) >> 3];
            const size_t r1 = (size_t)((wv + 8) * 16 + lrow) * D_ + kb + ks;
            a1.q[0] = xb4[(r1 + (size_t)half * 8) >> 3];
            a1.q[1] = xb4[(r1 + 16 + (size_t)half * 8) >> 3];
#pragma unroll
            for (int nt = 0; nt < 4; ++nt) {
                FragAB bb;   // B fragment: lane lrow = N column = feature row
                bb.q[0] = *(const uint4*)&sF[nt * 16 + lrow][ks + half * 8];
                bb.q[1] = *(const uint4*)&sF[nt * 16 + lrow][ks + 16 + half * 8];
                acc0[nt] = __builtin_amdgcn_wmma_f32_16x16x32_bf16(
                               false, a0.v, false, bb.v, (short)0, acc0[nt], false, false);
                acc1[nt] = __builtin_amdgcn_wmma_f32_16x16x32_bf16(
                               false, a1.v, false, bb.v, (short)0, acc1[nt], false, false);
            }
        }
        __syncthreads();
    }

    // ---- epilogue: per-row online-softmax partials over this block's 64 columns ----
    // C layout: VGPR j -> M = j + half*8 (lanes 0-15 / 16-31), N = lrow.
#pragma unroll
    for (int a = 0; a < 2; ++a) {
        v8f* acc = a ? acc1 : acc0;
        const int rbase = (wv + a * 8) * 16 + half * 8;
#pragma unroll
        for (int j = 0; j < 8; ++j) {
            float m = -INFINITY;
#pragma unroll
            for (int nt = 0; nt < 4; ++nt) m = fmaxf(m, acc[nt][j] * TEMP_INV);
#pragma unroll
            for (int off = 1; off < 16; off <<= 1)
                m = fmaxf(m, __shfl_xor(m, off, 32));
            float s = 0.f;
#pragma unroll
            for (int nt = 0; nt < 4; ++nt) s += __expf(acc[nt][j] * TEMP_INV - m);
#pragma unroll
            for (int off = 1; off < 16; off <<= 1)
                s += __shfl_xor(s, off, 32);
            if (lrow == 0) {
                const int row = rbase + j;
                pmax[(size_t)row * NBLK_ + blockIdx.x] = m;
                psum[(size_t)row * NBLK_ + blockIdx.x] = s;
            }
        }
    }
}

// ---------- 3) exact fp32 d[b] = x[b] . feat[tgt[b]] ----------
__global__ void __launch_bounds__(256) dot_own(
    const float* __restrict__ x, const float* __restrict__ feat,
    const int* __restrict__ tgt, float* __restrict__ dws)
{
    __shared__ float sred[256];
    const int b = blockIdx.x, t = threadIdx.x;
    const int tg = tgt[b];
    float acc = 0.f;
    for (int i = t; i < D_; i += 256)
        acc += x[(size_t)b * D_ + i] * feat[(size_t)tg * D_ + i];
    acc = block_reduce_sum(acc, sred);
    if (t == 0) dws[b] = acc;
}

// ---------- 4) combine per-block partials -> row LSE and CE contribution ----------
__global__ void __launch_bounds__(256) row_lse_kernel(
    const float* __restrict__ pmax, const float* __restrict__ psum,
    const float* __restrict__ dws, float* __restrict__ rowloss)
{
    __shared__ float sred[256];
    const int b = blockIdx.x, t = threadIdx.x;
    const float* pm = pmax + (size_t)b * NBLK_;
    const float* ps = psum + (size_t)b * NBLK_;
    float mx = -INFINITY;
    for (int i = t; i < NBLK_; i += 256) mx = fmaxf(mx, pm[i]);
    sred[t] = mx; __syncthreads();
    for (int s = 128; s > 0; s >>= 1) {
        if (t < s) sred[t] = fmaxf(sred[t], sred[t + s]);
        __syncthreads();
    }
    mx = sred[0]; __syncthreads();
    float sum = 0.f;
    for (int i = t; i < NBLK_; i += 256) sum += ps[i] * __expf(pm[i] - mx);
    sum = block_reduce_sum(sum, sred);
    if (t == 0) {
        const float lse = mx + __logf(sum);
        rowloss[b] = lse - dws[b] * TEMP_INV;   // logit[target] = d[b]/TEMP
    }
}

// ---------- 5) init min-key / sel ----------
__global__ void init_minsel(unsigned* __restrict__ minkey, unsigned* __restrict__ sel) {
    const int i = blockIdx.x * 256 + threadIdx.x;
    if (i < N_) { minkey[i] = 0xFFFFFFFFu; sel[i] = B_; }
}

// ---------- 6) loss = mean(rowloss) ----------
__global__ void __launch_bounds__(256) loss_sum(
    const float* __restrict__ rowloss, float* __restrict__ out)
{
    __shared__ float sred[256];
    float v = rowloss[threadIdx.x];
    v = block_reduce_sum(v, sred);
    if (threadIdx.x == 0) out[0] = v / (float)B_;
}

// ---------- 7) scatter-min of d per cluster ----------
__global__ void scatter_min(const float* __restrict__ dws, const int* __restrict__ tgt,
                            unsigned* __restrict__ minkey)
{
    const int b = threadIdx.x;
    atomicMin(&minkey[tgt[b]], fkey(dws[b]));
}

// ---------- 8) first batch-index minimizer per cluster ----------
__global__ void scatter_sel(const float* __restrict__ dws, const int* __restrict__ tgt,
                            const unsigned* __restrict__ minkey, unsigned* __restrict__ sel)
{
    const int b = threadIdx.x;
    const int t = tgt[b];
    if (fkey(dws[b]) <= minkey[t]) atomicMin(&sel[t], (unsigned)b);
}

// ---------- 9) EMA + renorm for chosen rows ----------
__global__ void __launch_bounds__(256) ema_update(
    const float* __restrict__ x, const float* __restrict__ feat,
    const int* __restrict__ tgt, const unsigned* __restrict__ sel,
    float* __restrict__ outfeat)
{
    __shared__ float se[D_];
    __shared__ float sred[256];
    const int b = blockIdx.x, t = threadIdx.x;
    const int tg = tgt[b];
    if (sel[tg] != (unsigned)b) return;   // uniform across block
    float ss = 0.f;
    for (int i = t; i < D_; i += 256) {
        float e = MOM * feat[(size_t)tg * D_ + i] + (1.0f - MOM) * x[(size_t)b * D_ + i];
        se[i] = e; ss += e * e;
    }
    ss = block_reduce_sum(ss, sred);
    const float scale = 1.0f / fmaxf(sqrtf(ss), EPS_);
    for (int i = t; i < D_; i += 256)
        outfeat[(size_t)tg * D_ + i] = se[i] * scale;
}

// ---------- launcher ----------
extern "C" void kernel_launch(void* const* d_in, const int* in_sizes, int n_in,
                              void* d_out, int out_size, void* d_ws, size_t ws_size,
                              hipStream_t stream) {
    const float* inputs  = (const float*)d_in[0];   // [B,D]
    const int*   targets = (const int*)d_in[1];     // [B]
    const float* feats   = (const float*)d_in[2];   // [N,D]

    float* out     = (float*)d_out;                 // [1 + N*D]
    float* outfeat = out + 1;

    char* w = (char*)d_ws;
    float*    x       = (float*)(w);                                    // 2 MB
    __bf16*   xb      = (__bf16*)(w + (size_t)(2 << 20));               // 1 MB
    float*    pmax    = (float*)(w + (size_t)(4 << 20));                // 1 MB
    float*    psum    = (float*)(w + (size_t)(5 << 20));                // 1 MB
    float*    dws     = (float*)(w + (size_t)(6 << 20));                // 1 KB
    float*    rowloss = (float*)(w + (size_t)(6 << 20) + 4096);         // 1 KB
    unsigned* minkey  = (unsigned*)(w + (size_t)(6 << 20) + 8192);      // 256 KB
    unsigned* sel     = (unsigned*)(w + (size_t)(6 << 20) + 8192 + N_ * 4);

    normalize_inputs<<<B_, 256, 0, stream>>>(inputs, x, xb);
    gemm_fused<<<NBLK_, 256, 0, stream>>>(feats, xb, pmax, psum, outfeat);
    dot_own<<<B_, 256, 0, stream>>>(x, feats, targets, dws);
    row_lse_kernel<<<B_, 256, 0, stream>>>(pmax, psum, dws, rowloss);
    init_minsel<<<N_ / 256, 256, 0, stream>>>(minkey, sel);
    loss_sum<<<1, 256, 0, stream>>>(rowloss, out);
    scatter_min<<<1, B_, 0, stream>>>(dws, targets, minkey);
    scatter_sel<<<1, B_, 0, stream>>>(dws, targets, minkey, sel);
    ema_update<<<B_, 256, 0, stream>>>(x, feats, targets, sel, outfeat);
}
